// GCN_48198122996027
// MI455X (gfx1250) — matrix operand order repaired and checked by reading, not compiled, not run on hardware
//
#include <hip/hip_runtime.h>
#include <hip/hip_bf16.h>

#define N_NODES 50000
#define N_EDGES 800000
#define FDIM    64
#define H2DIM   128
#define NGRAPH  64
#define NCLS    2
#define KCH     5

typedef __attribute__((ext_vector_type(16))) _Float16 v16h;
typedef __attribute__((ext_vector_type(8)))  _Float16 v8h;
typedef __attribute__((ext_vector_type(8)))  float    v8f;

static inline int gridFor(long long n) { return (int)((n + 255) / 256); }

struct T5 { const float* t0; const float* t1; const float* t2; const float* t3; const float* t4; };

// ---------------- elementwise / init kernels ----------------

__global__ void k_zero(float* __restrict__ p, int n) {
    int i = blockIdx.x * blockDim.x + threadIdx.x;
    if (i < n) p[i] = 0.0f;
}

__global__ void k_neg(float* __restrict__ d, const float* __restrict__ s, int n) {
    int i = blockIdx.x * blockDim.x + threadIdx.x;
    if (i < n) d[i] = -s[i];
}

// Wt[k][n][kk] = (f16) W[k][kk][n]  -- weight transpose+convert so WMMA B tiles are
// contiguous 16-half runs (two b128 loads per tile).
__global__ void k_wcvt(_Float16* __restrict__ Wt, const float* __restrict__ W,
                       int Fout, int n) {
    int i = blockIdx.x * blockDim.x + threadIdx.x;
    if (i >= n) return;
    int kk = i % FDIM;
    int col = (i / FDIM) % Fout;
    int k = i / (FDIM * Fout);
    Wt[i] = (_Float16)W[((size_t)k * FDIM + kk) * Fout + col];
}

// ---------------- graph normalization ----------------

__global__ void k_deg(float* __restrict__ deg, const int* __restrict__ src,
                      const float* __restrict__ w, int e) {
    int i = blockIdx.x * blockDim.x + threadIdx.x;
    if (i < e) atomicAdd(deg + src[i], w[i]);
}

__global__ void k_rsqrt(float* __restrict__ deg, int n) {
    int i = blockIdx.x * blockDim.x + threadIdx.x;
    if (i < n) {
        float v = deg[i];
        deg[i] = (v > 0.0f) ? rsqrtf(v) : 0.0f;
    }
}

__global__ void k_norm(float* __restrict__ norm, const float* __restrict__ dis,
                       const int* __restrict__ src, const int* __restrict__ dst,
                       const float* __restrict__ w, int e) {
    int i = blockIdx.x * blockDim.x + threadIdx.x;
    if (i < e) norm[i] = -dis[src[i]] * w[i] * dis[dst[i]];
}

// ---------------- sparse propagation: out[dst] += scale*norm*z[src] ----------------
// 64 lanes per edge (lane = feature); coalesced 256B gather + f32 atomic scatter.

__global__ __launch_bounds__(256)
void k_prop(const float* __restrict__ z, const int* __restrict__ src,
            const int* __restrict__ dst, const float* __restrict__ norm,
            float* __restrict__ out, int e, float scale) {
    long long t = (long long)blockIdx.x * blockDim.x + threadIdx.x;
    int ei = (int)(t >> 6);
    if (ei >= e) return;
    int f = (int)(t & 63);
    float v = scale * norm[ei] * z[(size_t)src[ei] * FDIM + f];
    atomicAdd(out + (size_t)dst[ei] * FDIM + f, v);
}

// ---------------- fused 5-way WMMA GEMM ----------------
// Hout[N x Fout] = relu( bias + sum_{k=0..4} T_k[N x 64] @ W[k][64 x Fout] )
// One wave per 16x16 output tile; 10 chained v_wmma_f32_16x16x32_f16, register-resident
// accumulator, single store pass (no RMW). Wt is pre-transposed f16 [k][col][kin].

__global__ __launch_bounds__(256)
void k_gemm5(T5 p, const _Float16* __restrict__ Wt, const float* __restrict__ bias,
             float* __restrict__ Y, int rowTiles, int Fout) {
    int wid = blockIdx.x * 8 + (threadIdx.x >> 5);
    int colTiles = Fout >> 4;
    if (wid >= rowTiles * colTiles) return;   // wave-uniform: EXEC stays all-ones
    int tm = wid / colTiles;
    int tn = wid % colTiles;
    int lane = threadIdx.x & 31;
    int half = lane >> 4;
    int r    = lane & 15;

    const float* Ts[KCH] = {p.t0, p.t1, p.t2, p.t3, p.t4};
    v8f c = {};
#pragma unroll
    for (int k = 0; k < KCH; ++k) {
#pragma unroll
        for (int k0 = 0; k0 < FDIM; k0 += 32) {
            v16h a, b;
            // A tile 16x32 f16: lane=row(M=r), elem i -> K = 16*half + i
            const float* ap = Ts[k] + (size_t)(tm * 16 + r) * FDIM + k0 + half * 16;
#pragma unroll
            for (int i = 0; i < 8; ++i) {
                a[i]     = (_Float16)ap[i];
                a[i + 8] = (_Float16)ap[8 + i];
            }
            // B tile 32x16 f16: lane=col(N=r), elem i -> K = k0 + 16*half + i.
            // Wt row for this column is contiguous: two b128 loads.
            const _Float16* bp =
                Wt + ((size_t)k * Fout + tn * 16 + r) * FDIM + k0 + half * 16;
            v8h blo = *(const v8h*)bp;
            v8h bhi = *(const v8h*)(bp + 8);
#pragma unroll
            for (int i = 0; i < 8; ++i) {
                b[i]     = blo[i];
                b[i + 8] = bhi[i];
            }
            c = __builtin_amdgcn_wmma_f32_16x16x32_f16(
                    /*neg_a=*/false, a, /*neg_b=*/false, b,
                    /*c_mod=*/(short)0, c, /*reuse_a=*/false, /*reuse_b=*/false);
        }
    }
    // C/D layout: VGPR j -> row M = 8*half + j, col N = r. Fused bias + relu, pure stores.
    float bv = bias[tn * 16 + r];
    float* yp = Y + (size_t)(tm * 16 + half * 8) * Fout + tn * 16 + r;
#pragma unroll
    for (int j = 0; j < 8; ++j) yp[(size_t)j * Fout] = fmaxf(c[j] + bv, 0.0f);
}

// ---------------- pooling + head ----------------

__global__ void k_pool_sum(float* __restrict__ sums, const float* __restrict__ h,
                           const int* __restrict__ batch, int n) {
    int i = blockIdx.x * blockDim.x + threadIdx.x;   // i = node*128 + f
    if (i >= n) return;
    int node = i >> 7;
    int f = i & 127;
    atomicAdd(sums + (size_t)batch[node] * H2DIM + f, h[i]);
}

__global__ void k_pool_cnt(float* __restrict__ cnt, const int* __restrict__ batch, int n) {
    int i = blockIdx.x * blockDim.x + threadIdx.x;
    if (i < n) atomicAdd(cnt + batch[i], 1.0f);
}

__global__ void k_hg(float* __restrict__ hg, const float* __restrict__ sums,
                     const float* __restrict__ cnt, int n) {
    int i = blockIdx.x * blockDim.x + threadIdx.x;
    if (i < n) hg[i] = sums[i] / fmaxf(cnt[i >> 7], 1.0f);
}

__global__ void k_head(float* __restrict__ out, const float* __restrict__ hg,
                       const float* __restrict__ Wl, const float* __restrict__ bl) {
    int i = threadIdx.x;                 // 0 .. G*C-1
    if (i >= NGRAPH * NCLS) return;
    int g = i / NCLS, cidx = i % NCLS;
    float acc = bl[cidx];
    for (int f = 0; f < H2DIM; ++f)
        acc += hg[(size_t)g * H2DIM + f] * Wl[(size_t)f * NCLS + cidx];
    out[i] = acc;
}

// ---------------- host orchestration ----------------

static void cheb_layer(const float* Xin, int Fout, const _Float16* Wt, const float* bias,
                       float* T1, float* T2, float* T3, float* T4, float* Hout,
                       const int* src, const int* dst, const float* norm,
                       hipStream_t stream) {
    const int nT = N_NODES * FDIM;
    const long long nPropT = (long long)N_EDGES * 64;
    // Chebyshev basis: T1 = L x ;  T_{k} = 2 L T_{k-1} - T_{k-2}
    k_zero<<<gridFor(nT), 256, 0, stream>>>(T1, nT);
    k_prop<<<gridFor(nPropT), 256, 0, stream>>>(Xin, src, dst, norm, T1, N_EDGES, 1.0f);
    k_neg <<<gridFor(nT), 256, 0, stream>>>(T2, Xin, nT);
    k_prop<<<gridFor(nPropT), 256, 0, stream>>>(T1, src, dst, norm, T2, N_EDGES, 2.0f);
    k_neg <<<gridFor(nT), 256, 0, stream>>>(T3, T1, nT);
    k_prop<<<gridFor(nPropT), 256, 0, stream>>>(T2, src, dst, norm, T3, N_EDGES, 2.0f);
    k_neg <<<gridFor(nT), 256, 0, stream>>>(T4, T2, nT);
    k_prop<<<gridFor(nPropT), 256, 0, stream>>>(T3, src, dst, norm, T4, N_EDGES, 2.0f);
    // Fused: Hout = relu(bias + sum_k T_k @ W[k]), 10 WMMAs per tile in-registers
    int rowTiles = N_NODES / 16;                 // 3125, exact
    int waves = rowTiles * (Fout / 16);
    T5 p{Xin, T1, T2, T3, T4};
    k_gemm5<<<(waves + 7) / 8, 256, 0, stream>>>(p, Wt, bias, Hout, rowTiles, Fout);
}

extern "C" void kernel_launch(void* const* d_in, const int* in_sizes, int n_in,
                              void* d_out, int out_size, void* d_ws, size_t ws_size,
                              hipStream_t stream) {
    const float* x     = (const float*)d_in[0];
    const int*   ei    = (const int*)d_in[1];
    const float* ea    = (const float*)d_in[2];
    const int*   batch = (const int*)d_in[3];
    const float* W1 = (const float*)d_in[4];
    const float* b1 = (const float*)d_in[5];
    const float* W2 = (const float*)d_in[6];
    const float* b2 = (const float*)d_in[7];
    const float* W3 = (const float*)d_in[8];
    const float* b3 = (const float*)d_in[9];
    const float* Wl = (const float*)d_in[10];
    const float* bl = (const float*)d_in[11];

    const int* src = ei;             // edge_index[0]
    const int* dst = ei + N_EDGES;   // edge_index[1]

    // workspace carve (256B aligned)
    size_t off = 0;
    char* base = (char*)d_ws;
    auto carve = [&](size_t bytes) -> char* {
        char* p = base + off;
        off += (bytes + 255) & ~(size_t)255;
        return p;
    };
    float* deg  = (float*)carve((size_t)N_NODES * 4);
    float* norm = (float*)carve((size_t)N_EDGES * 4);
    float* T1   = (float*)carve((size_t)N_NODES * FDIM * 4);
    float* T2   = (float*)carve((size_t)N_NODES * FDIM * 4);
    float* T3   = (float*)carve((size_t)N_NODES * FDIM * 4);
    float* T4   = (float*)carve((size_t)N_NODES * FDIM * 4);
    float* Ha   = (float*)carve((size_t)N_NODES * FDIM * 4);
    float* Hb   = (float*)carve((size_t)N_NODES * FDIM * 4);
    float* Hc   = (float*)carve((size_t)N_NODES * H2DIM * 4);
    float* sums = (float*)carve((size_t)NGRAPH * H2DIM * 4);
    float* cnt  = (float*)carve((size_t)NGRAPH * 4);
    _Float16* Wt1 = (_Float16*)carve((size_t)KCH * FDIM * FDIM * 2);
    _Float16* Wt2 = (_Float16*)carve((size_t)KCH * FDIM * FDIM * 2);
    _Float16* Wt3 = (_Float16*)carve((size_t)KCH * FDIM * H2DIM * 2);
    (void)ws_size; (void)in_sizes; (void)n_in; (void)out_size;

    // weight transpose+convert (tiny, once per call)
    k_wcvt<<<gridFor(KCH * FDIM * FDIM), 256, 0, stream>>>(Wt1, W1, FDIM,
                                                           KCH * FDIM * FDIM);
    k_wcvt<<<gridFor(KCH * FDIM * FDIM), 256, 0, stream>>>(Wt2, W2, FDIM,
                                                           KCH * FDIM * FDIM);
    k_wcvt<<<gridFor(KCH * FDIM * H2DIM), 256, 0, stream>>>(Wt3, W3, H2DIM,
                                                            KCH * FDIM * H2DIM);

    // symmetric normalization of scaled Laplacian (off-diagonal part)
    k_zero <<<gridFor(N_NODES), 256, 0, stream>>>(deg, N_NODES);
    k_deg  <<<gridFor(N_EDGES), 256, 0, stream>>>(deg, src, ea, N_EDGES);
    k_rsqrt<<<gridFor(N_NODES), 256, 0, stream>>>(deg, N_NODES);
    k_norm <<<gridFor(N_EDGES), 256, 0, stream>>>(norm, deg, src, dst, ea, N_EDGES);

    // three Chebyshev layers (bases then one fused WMMA GEMM each)
    cheb_layer(x,  FDIM,  Wt1, b1, T1, T2, T3, T4, Ha, src, dst, norm, stream);
    cheb_layer(Ha, FDIM,  Wt2, b2, T1, T2, T3, T4, Hb, src, dst, norm, stream);
    cheb_layer(Hb, H2DIM, Wt3, b3, T1, T2, T3, T4, Hc, src, dst, norm, stream);

    // global mean pool + linear head
    k_zero<<<gridFor(NGRAPH * H2DIM), 256, 0, stream>>>(sums, NGRAPH * H2DIM);
    k_zero<<<gridFor(NGRAPH), 256, 0, stream>>>(cnt, NGRAPH);
    k_pool_sum<<<gridFor((long long)N_NODES * H2DIM), 256, 0, stream>>>(sums, Hc, batch,
                                                                        N_NODES * H2DIM);
    k_pool_cnt<<<gridFor(N_NODES), 256, 0, stream>>>(cnt, batch, N_NODES);

    float* out_logits = (float*)d_out;                    // (out, hg) tuple order
    float* out_hg     = (float*)d_out + NGRAPH * NCLS;
    k_hg  <<<gridFor(NGRAPH * H2DIM), 256, 0, stream>>>(out_hg, sums, cnt, NGRAPH * H2DIM);
    k_head<<<1, 128, 0, stream>>>(out_logits, out_hg, Wl, bl);
}
